// DppMicroAttentionLayer_5299989643987
// MI455X (gfx1250) — matrix operand order repaired and checked by reading, not compiled, not run on hardware
//
#include <hip/hip_runtime.h>
#include <hip/hip_bf16.h>

// ---------------------------------------------------------------------------
// Shapes (compile-time constants from the reference)
// ---------------------------------------------------------------------------
#define BB 16
#define TT 1024
#define SS 2048
#define CC 256
#define EE 256
#define NSAMP 103          // ceil(2048/20) sampled key rows (stride 20, start 0)
#define MS 20              // MAX_SAMPLES
#define RSQRT_HALF 0.70710678118654752f

typedef __attribute__((ext_vector_type(16))) __bf16 v16bf;
typedef __attribute__((ext_vector_type(8)))  float  v8f;

// ---------------------------------------------------------------------------
// WMMA helpers (bf16 inputs, f32 accumulate) -- wave32, 16x16 tile per WMMA
// ---------------------------------------------------------------------------
__device__ __forceinline__ v8f wmma_bf16(v16bf a, v16bf b, v8f c) {
  return __builtin_amdgcn_wmma_f32_16x16x32_bf16(
      /*neg_a=*/false, a, /*neg_b=*/false, b,
      /*c_mod=*/(short)0, c, /*reuse_a=*/false, /*reuse_b=*/false);
}

// A tile: 16 (rows m) x 32 (k), row-major in memory with leading dim ld.
// 16-bit A layout: lane L holds row m=L%16; half=L/16 selects K groups
//   element j: k = half*16 + j (contiguous, matches B loaders)
__device__ __forceinline__ v16bf load_a_rowmajor(const float* __restrict__ p, int ld) {
  int lane = threadIdx.x & 31;
  int m    = lane & 15;
  int half = lane >> 4;
  const float* row = p + (size_t)m * ld;
  v16bf out;
#pragma unroll
  for (int j = 0; j < 16; ++j) {
    int i = j >> 1;
    int k = (half << 4) + j; (void)i;
    out[j] = (__bf16)row[k];
  }
  return out;
}

// B tile: 32 (k) x 16 (n), element [k][n] at p[k*ld + n] (k-major memory).
// B order used here (contiguous, same as A): lanes 0-15 hold K=0..15 (n=lane), lanes 16-31 K=16..31.
__device__ __forceinline__ v16bf load_b_kmajor(const float* __restrict__ p, int ld) {
  int lane  = threadIdx.x & 31;
  int n     = lane & 15;
  int kbase = (lane >> 4) << 4;
  v16bf out;
#pragma unroll
  for (int j = 0; j < 16; ++j) out[j] = (__bf16)p[(size_t)(kbase + j) * ld + n];
  return out;
}

// B tile where memory is n-major: element [k][n] at p[n*ld + k] (weights W[n][k]).
__device__ __forceinline__ v16bf load_b_nmajor(const float* __restrict__ p, int ld) {
  int lane  = threadIdx.x & 31;
  int n     = lane & 15;
  int kbase = (lane >> 4) << 4;
  const float* row = p + (size_t)n * ld + kbase;
  v16bf out;
#pragma unroll
  for (int j = 0; j < 16; ++j) out[j] = (__bf16)row[j];
  return out;
}

// C/D 16x16 f32 layout: vgpr r holds row m = (lane/16)*8 + r, col n = lane%16
__device__ __forceinline__ void store_c(float* __restrict__ p, int ld, v8f c, float scale) {
  int lane  = threadIdx.x & 31;
  int n     = lane & 15;
  int mbase = (lane >> 4) << 3;
#pragma unroll
  for (int r = 0; r < 8; ++r) p[(size_t)(mbase + r) * ld + n] = c[r] * scale;
}

// ---------------------------------------------------------------------------
// K1: h = (x @ W_in^T + b_in + te) * sqrt(0.5)   [16384 x 256], K=256
// Register-blocked: each wave computes a 16x64 strip (4 accumulators).
// ---------------------------------------------------------------------------
__global__ void k_gemm_in(const float* __restrict__ x, const float* __restrict__ W_in,
                          const float* __restrict__ b_in, const float* __restrict__ te,
                          float* __restrict__ h) {
  int wave = threadIdx.x >> 5;
  int tile = blockIdx.x * 8 + wave;          // (BT/16)*(E/64) = 1024*4 strips
  int tn4 = tile & 3;
  int tm  = tile >> 2;
  const float* A  = x    + (size_t)tm * 16 * CC;
  const float* Bp = W_in + (size_t)tn4 * 64 * CC;   // W_in[e][c]: n-major over k=c
  v8f acc[4] = {{}, {}, {}, {}};
  for (int ks = 0; ks < CC; ks += 32) {
    v16bf a = load_a_rowmajor(A + ks, CC);
#pragma unroll
    for (int u = 0; u < 4; ++u) {
      v16bf bt = load_b_nmajor(Bp + (size_t)u * 16 * CC + ks, CC);
      acc[u] = wmma_bf16(a, bt, acc[u]);
    }
  }
  int lane = threadIdx.x & 31;
  int n = lane & 15, mbase = (lane >> 4) << 3;
#pragma unroll
  for (int u = 0; u < 4; ++u) {
    int e = tn4 * 64 + u * 16 + n;
    float bias = b_in[e];
    const float* tep = te + (size_t)tm * 16 * EE + e;
    float*       hp  = h  + (size_t)tm * 16 * EE + e;
#pragma unroll
    for (int r = 0; r < 8; ++r) {
      size_t m = (size_t)(mbase + r);
      hp[m * EE] = (acc[u][r] + bias + tep[m * EE]) * RSQRT_HALF;
    }
  }
}

// ---------------------------------------------------------------------------
// K2a: scores = h @ enc_keys  -> written into attn region of d_out (scratch)
// Register-blocked 16x64 strip per wave.
// ---------------------------------------------------------------------------
__global__ void k_gemm_scores(const float* __restrict__ h, const float* __restrict__ enc_keys,
                              float* __restrict__ scores) {
  int wave = threadIdx.x >> 5;
  int tile = blockIdx.x * 8 + wave;          // (BT/16)*(S/64) = 1024*32 strips
  int tn4 = tile & 31;
  int tmg = tile >> 5;                        // global row tile 0..1023
  int b   = tmg >> 6;                         // 64 row tiles per batch
  const float* A  = h + (size_t)tmg * 16 * EE;
  const float* Kb = enc_keys + (size_t)b * EE * SS + (size_t)tn4 * 64;  // [e][s]
  v8f acc[4] = {{}, {}, {}, {}};
  for (int ks = 0; ks < EE; ks += 32) {
    v16bf a = load_a_rowmajor(A + ks, EE);
#pragma unroll
    for (int u = 0; u < 4; ++u) {
      v16bf bt = load_b_kmajor(Kb + (size_t)ks * SS + u * 16, SS);
      acc[u] = wmma_bf16(a, bt, acc[u]);
    }
  }
  float* Cp = scores + (size_t)tmg * 16 * SS + (size_t)tn4 * 64;
#pragma unroll
  for (int u = 0; u < 4; ++u) store_c(Cp + u * 16, SS, acc[u], 1.0f);
}

// ---------------------------------------------------------------------------
// K2b: in-place row softmax over S=2048 (one block per row, 256 threads)
// ---------------------------------------------------------------------------
__global__ void k_softmax_rows(float* __restrict__ attn) {
  __shared__ float red[256];
  float* p = attn + (size_t)blockIdx.x * SS;
  int tid = threadIdx.x;
  float mx = -3.402823466e38f;
#pragma unroll
  for (int i = 0; i < 8; ++i) mx = fmaxf(mx, p[tid + i * 256]);
  red[tid] = mx; __syncthreads();
  for (int off = 128; off > 0; off >>= 1) {
    if (tid < off) red[tid] = fmaxf(red[tid], red[tid + off]);
    __syncthreads();
  }
  float m = red[0]; __syncthreads();
  float vals[8], sum = 0.0f;
#pragma unroll
  for (int i = 0; i < 8; ++i) { vals[i] = __expf(p[tid + i * 256] - m); sum += vals[i]; }
  red[tid] = sum; __syncthreads();
  for (int off = 128; off > 0; off >>= 1) {
    if (tid < off) red[tid] += red[tid + off];
    __syncthreads();
  }
  float inv = 1.0f / red[0];
#pragma unroll
  for (int i = 0; i < 8; ++i) p[tid + i * 256] = vals[i] * inv;
}

// ---------------------------------------------------------------------------
// x_mean[b][s] = mean_t attn[b][t][s]   (deterministic, no atomics)
// ---------------------------------------------------------------------------
__global__ void k_xmean(const float* __restrict__ attn, float* __restrict__ xmean) {
  int idx = blockIdx.x * 256 + threadIdx.x;   // b*S + s, 16*2048 total
  int b = idx >> 11, s = idx & 2047;
  const float* p = attn + (size_t)b * TT * SS + s;
  float acc = 0.0f;
  for (int t = 0; t < TT; ++t) acc += p[(size_t)t * SS];
  xmean[idx] = acc * (1.0f / (float)TT);
}

// ---------------------------------------------------------------------------
// DPP: build L (103x103) and run 20-step greedy selection; mu = sorted indices
// One block (128 threads) per batch. mu values are RAW indices (reference quirk).
// ---------------------------------------------------------------------------
__global__ void k_dpp(const float* __restrict__ enc_values, const float* __restrict__ xmean,
                      float* __restrict__ mu) {
  __shared__ float L[NSAMP * NSAMP];
  __shared__ float Cm[NSAMP * MS];
  __shared__ float Dv[NSAMP];
  __shared__ float maskv[NSAMP];
  __shared__ float aa[NSAMP];
  __shared__ float innorm[NSAMP];
  __shared__ float cj[MS];
  __shared__ int   selJ[MS];
  int b = blockIdx.x, tid = threadIdx.x;
  const float* V = enc_values + (size_t)b * SS * EE;

  for (int i = tid; i < NSAMP; i += 128) {
    const float* v = V + (size_t)i * 20 * EE;
    float s2 = 0.0f;
    for (int e = 0; e < EE; ++e) s2 += v[e] * v[e];
    innorm[i] = rsqrtf(s2);
    aa[i]     = xmean[b * SS + i * 20];
    maskv[i]  = 1.0f;
    for (int k = 0; k < MS; ++k) Cm[i * MS + k] = 0.0f;
  }
  __syncthreads();

  for (int pidx = tid; pidx < NSAMP * NSAMP; pidx += 128) {
    int i = pidx / NSAMP, j = pidx % NSAMP;
    const float* vi = V + (size_t)i * 20 * EE;
    const float* vj = V + (size_t)j * 20 * EE;
    float d = 0.0f;
    for (int e = 0; e < EE; ++e) d += vi[e] * vj[e];
    L[pidx] = (aa[i] * aa[j]) * (d * innorm[i] * innorm[j]);
  }
  __syncthreads();
  if (tid < NSAMP) Dv[tid] = L[tid * NSAMP + tid];
  __syncthreads();

  if (tid == 0) {
    int bj = 0; float bv = -3.402823466e38f;
    for (int n = 0; n < NSAMP; ++n) {
      float v = logf(Dv[n] * maskv[n]);
      if (v > bv) { bv = v; bj = n; }
    }
    selJ[0] = bj; maskv[bj] = 0.0f;
  }
  __syncthreads();

  for (int t = 1; t < MS; ++t) {
    int J = selJ[t - 1];
    if (tid < MS) cj[tid] = Cm[J * MS + tid];
    __syncthreads();
    float DJ = Dv[J];
    for (int n = tid; n < NSAMP; n += 128) {
      float cd = 0.0f;
      for (int k = 0; k < t; ++k) cd += Cm[n * MS + k] * cj[k];
      float ev = (L[J * NSAMP + n] - cd) / DJ * maskv[n];
      Cm[n * MS + t] = ev;
      Dv[n] -= ev * ev;
    }
    __syncthreads();
    if (tid == 0) {
      int bj = 0; float bv = -3.402823466e38f;
      for (int n = 0; n < NSAMP; ++n) {
        float v = logf(Dv[n] * maskv[n]);
        if (v > bv) { bv = v; bj = n; }
      }
      selJ[t] = bj; maskv[bj] = 0.0f;
    }
    __syncthreads();
  }

  if (tid == 0) {
    int s[MS];
    for (int i = 0; i < MS; ++i) s[i] = selJ[i];
    for (int i = 1; i < MS; ++i) {             // insertion sort ascending
      int key = s[i], j = i - 1;
      while (j >= 0 && s[j] > key) { s[j + 1] = s[j]; --j; }
      s[j + 1] = key;
    }
    for (int i = 0; i < MS; ++i) mu[b * MS + i] = (float)s[i];
  }
}

// ---------------------------------------------------------------------------
// dist = softmax_s( sum_k N(s; mu_k, sigma) ) ; klpart[b] = sum_s xm*(log xm - dist)
// ---------------------------------------------------------------------------
__global__ void k_gm_kl(const float* __restrict__ mu, const float* __restrict__ xmean,
                        float* __restrict__ klpart) {
  __shared__ float red[256];
  __shared__ float msh[MS];
  int b = blockIdx.x, tid = threadIdx.x;
  if (tid < MS) msh[tid] = mu[b * MS + tid];
  __syncthreads();
  const float c0     = 0.13298076013381091f;     // 1/(sqrt(2*pi)*3)
  const float inv2s2 = 1.0f / 18.0f;             // 1/(2*sigma^2), sigma=3
  float g[8];
  float mx = -3.402823466e38f;
#pragma unroll
  for (int i = 0; i < 8; ++i) {
    int s = tid + i * 256;
    float acc = 0.0f;
    for (int k = 0; k < MS; ++k) {
      float z = (float)s - msh[k];
      acc += __expf(-z * z * inv2s2);
    }
    g[i] = acc * c0;
    mx = fmaxf(mx, g[i]);
  }
  red[tid] = mx; __syncthreads();
  for (int off = 128; off > 0; off >>= 1) {
    if (tid < off) red[tid] = fmaxf(red[tid], red[tid + off]);
    __syncthreads();
  }
  float m = red[0]; __syncthreads();
  float sum = 0.0f;
#pragma unroll
  for (int i = 0; i < 8; ++i) { g[i] = __expf(g[i] - m); sum += g[i]; }
  red[tid] = sum; __syncthreads();
  for (int off = 128; off > 0; off >>= 1) {
    if (tid < off) red[tid] += red[tid + off];
    __syncthreads();
  }
  float inv = 1.0f / red[0]; __syncthreads();
  float kl = 0.0f;
  const float* xm = xmean + b * SS;
#pragma unroll
  for (int i = 0; i < 8; ++i) {
    int s = tid + i * 256;
    float xv = xm[s];
    kl += xv * (logf(xv) - g[i] * inv);
  }
  red[tid] = kl; __syncthreads();
  for (int off = 128; off > 0; off >>= 1) {
    if (tid < off) red[tid] += red[tid + off];
    __syncthreads();
  }
  if (tid == 0) klpart[b] = red[0];
}

__global__ void k_kl_final(const float* __restrict__ klpart, float* __restrict__ out_kl) {
  if (threadIdx.x == 0 && blockIdx.x == 0) {
    float s = 0.0f;
    for (int b = 0; b < BB; ++b) s += klpart[b];
    out_kl[0] = -s;
  }
}

// ---------------------------------------------------------------------------
// K3: tmp = (attn @ enc_values) * sqrt(S)   [16384 x 256], K=2048
// Register-blocked 16x64 strip per wave.
// ---------------------------------------------------------------------------
__global__ void k_gemm_av(const float* __restrict__ attn, const float* __restrict__ enc_values,
                          float* __restrict__ tmp) {
  int wave = threadIdx.x >> 5;
  int tile = blockIdx.x * 8 + wave;          // (BT/16)*(E/64) = 1024*4 strips
  int tn4 = tile & 3;
  int tmg = tile >> 2;
  int b   = tmg >> 6;
  const float* A = attn + (size_t)tmg * 16 * SS;
  const float* V = enc_values + (size_t)b * SS * EE + (size_t)tn4 * 64;  // [s][e]
  v8f acc[4] = {{}, {}, {}, {}};
  for (int ks = 0; ks < SS; ks += 32) {
    v16bf a = load_a_rowmajor(A + ks, SS);
#pragma unroll
    for (int u = 0; u < 4; ++u) {
      v16bf bt = load_b_kmajor(V + (size_t)ks * EE + u * 16, EE);
      acc[u] = wmma_bf16(a, bt, acc[u]);
    }
  }
  const float scale = 45.254833995939045f;    // 2048 * sqrt(1/2048) = sqrt(2048)
  float* Cp = tmp + (size_t)tmg * 16 * EE + (size_t)tn4 * 64;
#pragma unroll
  for (int u = 0; u < 4; ++u) store_c(Cp + u * 16, EE, acc[u], scale);
}

// ---------------------------------------------------------------------------
// K4: out = (tmp @ W_out^T + b_out + x) * sqrt(0.5)   [16384 x 256], K=256
// Register-blocked 16x64 strip per wave.
// ---------------------------------------------------------------------------
__global__ void k_gemm_out(const float* __restrict__ tmp, const float* __restrict__ W_out,
                           const float* __restrict__ b_out, const float* __restrict__ x,
                           float* __restrict__ out) {
  int wave = threadIdx.x >> 5;
  int tile = blockIdx.x * 8 + wave;          // (BT/16)*(C/64) = 1024*4 strips
  int tn4 = tile & 3;
  int tmg = tile >> 2;
  const float* A  = tmp   + (size_t)tmg * 16 * EE;
  const float* Bp = W_out + (size_t)tn4 * 64 * EE;   // W_out[c][e]: n-major over k=e
  v8f acc[4] = {{}, {}, {}, {}};
  for (int ks = 0; ks < EE; ks += 32) {
    v16bf a = load_a_rowmajor(A + ks, EE);
#pragma unroll
    for (int u = 0; u < 4; ++u) {
      v16bf bt = load_b_nmajor(Bp + (size_t)u * 16 * EE + ks, EE);
      acc[u] = wmma_bf16(a, bt, acc[u]);
    }
  }
  int lane = threadIdx.x & 31;
  int n = lane & 15, mbase = (lane >> 4) << 3;
#pragma unroll
  for (int u = 0; u < 4; ++u) {
    int c = tn4 * 64 + u * 16 + n;
    float bias = b_out[c];
    const float* xp = x   + (size_t)tmg * 16 * CC + c;
    float*       op = out + (size_t)tmg * 16 * CC + c;
#pragma unroll
    for (int r = 0; r < 8; ++r) {
      size_t m = (size_t)(mbase + r);
      op[m * CC] = (acc[u][r] + bias + xp[m * CC]) * RSQRT_HALF;
    }
  }
}

// ---------------------------------------------------------------------------
extern "C" void kernel_launch(void* const* d_in, const int* in_sizes, int n_in,
                              void* d_out, int out_size, void* d_ws, size_t ws_size,
                              hipStream_t stream) {
  (void)in_sizes; (void)n_in; (void)out_size; (void)ws_size;
  const float* x      = (const float*)d_in[0];
  const float* te     = (const float*)d_in[1];
  const float* enc_k  = (const float*)d_in[2];
  const float* enc_v  = (const float*)d_in[3];
  const float* W_in   = (const float*)d_in[4];
  const float* b_in   = (const float*)d_in[5];
  const float* W_out  = (const float*)d_in[6];
  const float* b_out  = (const float*)d_in[7];

  float* out  = (float*)d_out;                         // (B,T,C)
  float* attn = out + (size_t)BB * TT * CC;            // (B,T,S)
  float* okl  = attn + (size_t)BB * TT * SS;           // scalar -kl

  float* ws     = (float*)d_ws;
  float* h      = ws;                                  // 16384*256
  float* tmp    = h   + (size_t)BB * TT * EE;          // 16384*256
  float* xmean  = tmp + (size_t)BB * TT * EE;          // 16*2048
  float* mu     = xmean + (size_t)BB * SS;             // 16*20
  float* klpart = mu + (size_t)BB * MS;                // 16

  // 1) h = (x W_in^T + b_in + te) * sqrt(.5)
  k_gemm_in<<<(BB * TT / 16) * (EE / 64) / 8, 256, 0, stream>>>(x, W_in, b_in, te, h);
  // 2) scores into attn region, then in-place softmax
  k_gemm_scores<<<(BB * TT / 16) * (SS / 64) / 8, 256, 0, stream>>>(h, enc_k, attn);
  k_softmax_rows<<<BB * TT, 256, 0, stream>>>(attn);
  // 3) x_mean
  k_xmean<<<(BB * SS) / 256, 256, 0, stream>>>(attn, xmean);
  // 4) DPP greedy selection -> mu
  k_dpp<<<BB, 128, 0, stream>>>(enc_v, xmean, mu);
  // 5) gaussian mixture, softmax over S, KL partials, final scalar
  k_gm_kl<<<BB, 256, 0, stream>>>(mu, xmean, klpart);
  k_kl_final<<<1, 32, 0, stream>>>(klpart, okl);
  // 6) tmp = attn @ enc_values * sqrt(S)
  k_gemm_av<<<(BB * TT / 16) * (EE / 64) / 8, 256, 0, stream>>>(attn, enc_v, tmp);
  // 7) out = (tmp W_out^T + b_out + x) * sqrt(.5)
  k_gemm_out<<<(BB * TT / 16) * (CC / 64) / 8, 256, 0, stream>>>(tmp, W_out, b_out, x, out);
}
